// GPT2LMHeadModel_60120952209787
// MI455X (gfx1250) — matrix operand order repaired and checked by reading, not compiled, not run on hardware
//
#include <hip/hip_runtime.h>
#include <hip/hip_bf16.h>

typedef __attribute__((ext_vector_type(16))) __bf16 v16bf;
typedef __attribute__((ext_vector_type(8)))  float  v8f;
typedef __attribute__((ext_vector_type(4)))  unsigned int u32x4;
typedef __attribute__((ext_vector_type(8)))  unsigned int u32x8;
typedef __attribute__((ext_vector_type(8)))  int i32x8;
typedef __attribute__((ext_vector_type(4)))  int i32x4;

union Frag16 { v16bf h; u32x8 u; };   // 16 bf16 viewed as 8 packed dwords

#define LAYERS 4
#define DMODEL 768
#define NH     12
#define HDIM   64
#define VSZ    50257
#define VPAD   50304                 // VSZ rounded up to 128 (tile/alignment pad)
#define BBATCH 2
#define TSEQ   1024
#define ROWS   (BBATCH * TSEQ)       // 2048

#ifndef __has_builtin
#define __has_builtin(x) 0
#endif
#if __has_builtin(__builtin_amdgcn_tensor_load_to_lds) && \
    __has_builtin(__builtin_amdgcn_s_wait_tensorcnt)
#define USE_TDM 1
#else
#define USE_TDM 0
#endif

static __device__ __forceinline__ __bf16 tobf(float f) { return (__bf16)f; }

// ---------------------------------------------------------------------------
// Embedding: x[b,t,:] = wte[ids[b,t],:] + wpe[t,:]   (fp32 residual stream)
// ---------------------------------------------------------------------------
__global__ __launch_bounds__(256) void embed_kernel(
    const int* __restrict__ ids, const float* __restrict__ wte,
    const float* __restrict__ wpe, float* __restrict__ x)
{
    int row = blockIdx.x;
    int t   = row % TSEQ;
    int id  = ids[row];
    for (int d = threadIdx.x; d < DMODEL; d += 256)
        x[(size_t)row * DMODEL + d] =
            wte[(size_t)id * DMODEL + d] + wpe[(size_t)t * DMODEL + d];
}

// ---------------------------------------------------------------------------
// fp32 -> bf16 bulk convert (weights; every launch, deterministic)
// ---------------------------------------------------------------------------
__global__ __launch_bounds__(256) void cvt_bf16_kernel(
    const float* __restrict__ in, __bf16* __restrict__ out, long long n4)
{
    long long i = (long long)blockIdx.x * 256 + threadIdx.x;   // n4 = n/4
    if (i < n4) {
        float4 v = *(const float4*)(in + i * 4);
        __bf16* o = out + i * 4;
        o[0] = tobf(v.x); o[1] = tobf(v.y); o[2] = tobf(v.z); o[3] = tobf(v.w);
    }
}

// ---------------------------------------------------------------------------
// wte (V,D) fp32 -> wteT (D, VPAD) bf16, zero-padded columns >= VSZ
// ---------------------------------------------------------------------------
__global__ __launch_bounds__(256) void transpose_wte_kernel(
    const float* __restrict__ wte, __bf16* __restrict__ wteT)
{
    __shared__ float tile[32][33];
    int v0 = blockIdx.x * 32, d0 = blockIdx.y * 32;
    int tx = threadIdx.x & 31, ty = threadIdx.x >> 5;    // 32 x 8
#pragma unroll
    for (int i = 0; i < 4; ++i) {
        int v = v0 + ty + i * 8;
        tile[ty + i * 8][tx] = (v < VSZ) ? wte[(size_t)v * DMODEL + d0 + tx] : 0.f;
    }
    __syncthreads();
#pragma unroll
    for (int i = 0; i < 4; ++i) {
        int d = d0 + ty + i * 8;
        wteT[(size_t)d * VPAD + v0 + tx] = tobf(tile[tx][ty + i * 8]);
    }
}

// ---------------------------------------------------------------------------
// LayerNorm: fp32 in, bf16 out (feeds WMMA GEMMs). One row per 256-thr block.
// ---------------------------------------------------------------------------
__global__ __launch_bounds__(256) void ln_kernel(
    const float* __restrict__ x, const float* __restrict__ g,
    const float* __restrict__ b, __bf16* __restrict__ o)
{
    __shared__ float red[256];
    int row = blockIdx.x;
    const float* xr = x + (size_t)row * DMODEL;

    float s = 0.f;
    for (int d = threadIdx.x; d < DMODEL; d += 256) s += xr[d];
    red[threadIdx.x] = s; __syncthreads();
    for (int st = 128; st > 0; st >>= 1) {
        if (threadIdx.x < st) red[threadIdx.x] += red[threadIdx.x + st];
        __syncthreads();
    }
    float mu = red[0] / DMODEL;
    __syncthreads();

    float v = 0.f;
    for (int d = threadIdx.x; d < DMODEL; d += 256) {
        float t = xr[d] - mu; v += t * t;
    }
    red[threadIdx.x] = v; __syncthreads();
    for (int st = 128; st > 0; st >>= 1) {
        if (threadIdx.x < st) red[threadIdx.x] += red[threadIdx.x + st];
        __syncthreads();
    }
    float inv = rsqrtf(red[0] / DMODEL + 1e-5f);

    for (int d = threadIdx.x; d < DMODEL; d += 256)
        o[(size_t)row * DMODEL + d] = tobf(g[d] * (xr[d] - mu) * inv + b[d]);
}

// ---------------------------------------------------------------------------
// TDM: 2-D tensor_load_to_lds descriptor (ISA ch.8); bf16 tiles, remaining
// extents so overhanging reads are hardware zero-filled.
// ---------------------------------------------------------------------------
#if USE_TDM
static __device__ __forceinline__ void tdm_load_2d(
    const __bf16* gsrc, unsigned lds_off, int rows, int cols,
    long long stride_elems, int rem_cols, int rem_rows)
{
    unsigned long long ga = (unsigned long long)(uintptr_t)gsrc;
    u32x4 g0;
    g0[0] = 1u;                                         // count=1, user D#
    g0[1] = lds_off;                                    // LDS byte address
    g0[2] = (unsigned)(ga & 0xFFFFFFFFull);             // global_addr[31:0]
    g0[3] = (unsigned)((ga >> 32) & 0x1FFFFFFull) | (2u << 30);  // [56:32]|type=2

    unsigned long long st = (unsigned long long)stride_elems;
    i32x8 g1;
    g1[0] = (int)(1u << 16);                            // data_size = 2 bytes
    g1[1] = (int)(((unsigned)rem_cols & 0xFFFFu) << 16);          // dim0[15:0]
    g1[2] = (int)((((unsigned)rem_cols >> 16) & 0xFFFFu) |
                  (((unsigned)rem_rows & 0xFFFFu) << 16));        // dim0 hi | dim1 lo
    g1[3] = (int)((((unsigned)rem_rows >> 16) & 0xFFFFu) |
                  ((unsigned)cols << 16));                        // dim1 hi | tile_dim0
    g1[4] = (int)((unsigned)rows & 0xFFFFu);                      // tile_dim1
    g1[5] = (int)(unsigned)(st & 0xFFFFFFFFull);                  // dim0_stride lo
    g1[6] = (int)(unsigned)((st >> 32) & 0xFFFFull);              // dim0_stride hi
    g1[7] = 0;
    i32x4 g2 = {0, 0, 0, 0};
    i32x4 g3 = {0, 0, 0, 0};
#if __clang_major__ >= 23
    i32x8 gz = {0, 0, 0, 0, 0, 0, 0, 0};
    __builtin_amdgcn_tensor_load_to_lds(g0, g1, g2, g3, gz, 0);
#else
    __builtin_amdgcn_tensor_load_to_lds(g0, g1, g2, g3, 0);
#endif
}
#endif

// ---------------------------------------------------------------------------
// bf16 WMMA GEMM: C = epi(A[M,K] @ W[K,N](ldb) + bias [+R])
//   EPI 0: fp32 C | 1: GELU->bf16 Cb | 2: +R->fp32 C | 3: bf16 Cb
// 256 thr = 8 waves (4x2), BM=128 BN=128 BK=32, wave tile 32x64.
// TDM double-buffered LDS tiles; sync bf16 copy fallback.
// ---------------------------------------------------------------------------
#define BM 128
#define BN 128
#define BK 32

static __device__ __forceinline__ void mma_tile(
    const __bf16* asb, const __bf16* bsb, v8f acc[2][4],
    int wm, int wn, int r, int hi)
{
    Frag16 af[2];
    v16bf bfr[4];
#pragma unroll
    for (int mt = 0; mt < 2; ++mt) {
        int rowb = (wm * 32 + mt * 16 + r) * BK;
#pragma unroll
        for (int e = 0; e < 8; ++e) {
            int kk = ((e < 4) ? 2 * e : 16 + 2 * (e - 4)) + 8 * hi;
            af[mt].u[e] = *(const unsigned*)(asb + rowb + kk);   // packed pair
        }
    }
#pragma unroll
    for (int nt = 0; nt < 4; ++nt) {
        int cb = wn * 64 + nt * 16 + r;
#pragma unroll
        for (int e = 0; e < 8; ++e) {
            int kk = ((e < 4) ? 2 * e : 16 + 2 * (e - 4)) + 8 * hi;
            bfr[nt][2 * e]     = bsb[kk * BN + cb];
            bfr[nt][2 * e + 1] = bsb[(kk + 1) * BN + cb];
        }
    }
#pragma unroll
    for (int mt = 0; mt < 2; ++mt)
#pragma unroll
        for (int nt = 0; nt < 4; ++nt)
            acc[mt][nt] = __builtin_amdgcn_wmma_f32_16x16x32_bf16(
                false, af[mt].h, false, bfr[nt], (short)0, acc[mt][nt],
                false, false);
}

template <int EPI>
__global__ __launch_bounds__(256) void gemm_wmma(
    const __bf16* __restrict__ A, const __bf16* __restrict__ W,
    const float* __restrict__ bias, const float* __restrict__ R,
    float* __restrict__ C, __bf16* __restrict__ Cb,
    int M, int N, int K, int ldb)
{
    __shared__ __align__(16) __bf16 As[2][BM * BK];
    __shared__ __align__(16) __bf16 Bs[2][BK * BN];

    int tid  = threadIdx.x;
    int lane = tid & 31, wave = tid >> 5;
    int wm = wave >> 1, wn = wave & 1;
    int r  = lane & 15, hi = lane >> 4;
    int m0 = blockIdx.y * BM;
    int n0 = blockIdx.x * BN;

    v8f acc[2][4];
#pragma unroll
    for (int i = 0; i < 2; ++i)
#pragma unroll
        for (int j = 0; j < 4; ++j)
#pragma unroll
            for (int e = 0; e < 8; ++e) acc[i][j][e] = 0.f;

    int nkb = K / BK;

#if USE_TDM
    if (wave == 0) {   // one wave drives the tensor DMA engine
        tdm_load_2d(A + (size_t)m0 * K,
                    (unsigned)(uintptr_t)&As[0][0], BM, BK, K, K, M - m0);
        tdm_load_2d(W + n0,
                    (unsigned)(uintptr_t)&Bs[0][0], BK, BN, ldb, ldb - n0, K);
    }
    for (int kb = 0; kb < nkb; ++kb) {
        int cur = kb & 1;
        __syncthreads();                    // all waves done reading buf cur^1
        if (wave == 0) {
            if (kb + 1 < nkb) {             // prefetch next tile pair
                tdm_load_2d(A + (size_t)m0 * K + (kb + 1) * BK,
                            (unsigned)(uintptr_t)&As[cur ^ 1][0],
                            BM, BK, K, K - (kb + 1) * BK, M - m0);
                tdm_load_2d(W + (size_t)(kb + 1) * BK * ldb + n0,
                            (unsigned)(uintptr_t)&Bs[cur ^ 1][0],
                            BK, BN, ldb, ldb - n0, K - (kb + 1) * BK);
                __builtin_amdgcn_s_wait_tensorcnt(2);   // tile kb complete
            } else {
                __builtin_amdgcn_s_wait_tensorcnt(0);
            }
        }
        __syncthreads();                    // tile kb visible to whole WG
        mma_tile(&As[cur][0], &Bs[cur][0], acc, wm, wn, r, hi);
    }
#else
    for (int kb = 0; kb < nkb; ++kb) {
        __syncthreads();
#pragma unroll
        for (int i = 0; i < 2; ++i) {       // A tile: 8KB, 16B chunks
            int c = tid + 256 * i;
            int row = c >> 2, c16 = c & 3;
            *(uint4*)(&As[0][0] + row * BK + c16 * 8) =
                *(const uint4*)(A + (size_t)(m0 + row) * K + kb * BK + c16 * 8);
        }
#pragma unroll
        for (int i = 0; i < 2; ++i) {       // B tile: 8KB (ldb pre-padded)
            int c = tid + 256 * i;
            int row = c >> 4, c16 = c & 15;
            *(uint4*)(&Bs[0][0] + row * BN + c16 * 8) =
                *(const uint4*)(W + (size_t)(kb * BK + row) * ldb + n0 + c16 * 8);
        }
        if (kb + 1 < nkb)
            __builtin_prefetch(W + (size_t)((kb + 1) * BK + (tid >> 3)) * ldb + n0, 0, 1);
        __syncthreads();
        mma_tile(&As[0][0], &Bs[0][0], acc, wm, wn, r, hi);
    }
#endif

#pragma unroll
    for (int mt = 0; mt < 2; ++mt)
#pragma unroll
        for (int nt = 0; nt < 4; ++nt)
#pragma unroll
            for (int e = 0; e < 8; ++e) {
                int m = m0 + wm * 32 + mt * 16 + e + 8 * hi;
                int n = n0 + wn * 64 + nt * 16 + r;
                if (n < N) {
                    float v = acc[mt][nt][e];
                    if (bias) v += bias[n];
                    if (EPI == 1) {
                        v = 0.5f * v * (1.f + erff(v * 0.70710678118f));
                        Cb[(size_t)m * N + n] = tobf(v);
                    } else if (EPI == 2) {
                        C[(size_t)m * N + n] = v + R[(size_t)m * N + n];
                    } else if (EPI == 3) {
                        Cb[(size_t)m * N + n] = tobf(v);
                    } else {
                        C[(size_t)m * N + n] = v;
                    }
                }
            }
}

// ---------------------------------------------------------------------------
// Causal attention (flash-style online softmax), WMMA for S=QK^T and O=PV.
// bf16 qkv in; K/V tiles staged via TDM into LDS, shared by all 4 waves
// (every wave walks the same key blocks). Block = 128 thr = 4 waves.
// ---------------------------------------------------------------------------
__global__ __launch_bounds__(128) void attn_kernel(
    const __bf16* __restrict__ qkv, __bf16* __restrict__ out)
{
    int qt = blockIdx.x, h = blockIdx.y, b = blockIdx.z;
    int wave = threadIdx.x >> 5, lane = threadIdx.x & 31;
    int r = lane & 15, hi = lane >> 4;
    int qbase = qt * 64 + wave * 16;
    const float scale = 0.125f;                 // 1/sqrt(HD), applied to S
    const size_t rs = 3 * DMODEL;

    const __bf16* Qp = qkv + (size_t)b * TSEQ * rs + h * HDIM;
    const __bf16* Kp = Qp + DMODEL;
    const __bf16* Vp = Qp + 2 * DMODEL;

    __shared__ __align__(16) __bf16 Kt[2][32 * HDIM];   // 32 keys x 64 dims
    __shared__ __align__(16) __bf16 Vt[2][32 * HDIM];
    __shared__ __bf16 Pb[4][16 * 32];

    // Q as two A-fragments (dims 0-31, 32-63), packed-dword loads
    Frag16 aq[2];
#pragma unroll
    for (int c = 0; c < 2; ++c) {
        const __bf16* qrow = Qp + (size_t)(qbase + r) * rs + c * 32;
#pragma unroll
        for (int e = 0; e < 8; ++e) {
            int kk = ((e < 4) ? 2 * e : 16 + 2 * (e - 4)) + 8 * hi;
            aq[c].u[e] = *(const unsigned*)(qrow + kk);
        }
    }

    float mrow[8], lrow[8];
    v8f o_acc[4];
#pragma unroll
    for (int e = 0; e < 8; ++e) { mrow[e] = -1e30f; lrow[e] = 0.f; }
#pragma unroll
    for (int nt = 0; nt < 4; ++nt)
#pragma unroll
        for (int e = 0; e < 8; ++e) o_acc[nt][e] = 0.f;

    int nkb = (qt + 1) * 2;     // uniform across waves -> barriers legal

#if USE_TDM
    if (wave == 0) {
        tdm_load_2d(Kp, (unsigned)(uintptr_t)&Kt[0][0], 32, HDIM, rs, HDIM, TSEQ);
        tdm_load_2d(Vp, (unsigned)(uintptr_t)&Vt[0][0], 32, HDIM, rs, HDIM, TSEQ);
    }
#endif
    for (int kb = 0; kb < nkb; ++kb) {
        int k0 = kb * 32;
#if USE_TDM
        int cur = kb & 1;
        __syncthreads();                    // waves done with buf cur^1
        if (wave == 0) {
            if (kb + 1 < nkb) {
                int k1 = (kb + 1) * 32;
                tdm_load_2d(Kp + (size_t)k1 * rs, (unsigned)(uintptr_t)&Kt[cur ^ 1][0],
                            32, HDIM, rs, HDIM, TSEQ - k1);
                tdm_load_2d(Vp + (size_t)k1 * rs, (unsigned)(uintptr_t)&Vt[cur ^ 1][0],
                            32, HDIM, rs, HDIM, TSEQ - k1);
                __builtin_amdgcn_s_wait_tensorcnt(2);
            } else {
                __builtin_amdgcn_s_wait_tensorcnt(0);
            }
        }
        __syncthreads();
#else
        const int cur = 0;
        __syncthreads();
#pragma unroll
        for (int i = 0; i < 2; ++i) {       // 4KB per buffer, 16B chunks
            int c = threadIdx.x + 128 * i;
            int row = c >> 3, c8 = c & 7;
            *(uint4*)(&Kt[0][0] + row * HDIM + c8 * 8) =
                *(const uint4*)(Kp + (size_t)(k0 + row) * rs + c8 * 8);
            *(uint4*)(&Vt[0][0] + row * HDIM + c8 * 8) =
                *(const uint4*)(Vp + (size_t)(k0 + row) * rs + c8 * 8);
        }
        __syncthreads();
#endif

        // ---- S = Q x K^T : 2 n-tiles (16 keys) x 2 k-steps (32 dims)
        v8f s[2];
#pragma unroll
        for (int nt = 0; nt < 2; ++nt) {
#pragma unroll
            for (int e = 0; e < 8; ++e) s[nt][e] = 0.f;
            const __bf16* krow = &Kt[cur][(nt * 16 + r) * HDIM];
#pragma unroll
            for (int c = 0; c < 2; ++c) {
                Frag16 bk;
#pragma unroll
                for (int e = 0; e < 8; ++e) {
                    int kk = ((e < 4) ? 2 * e : 16 + 2 * (e - 4)) + 8 * hi + c * 32;
                    bk.u[e] = *(const unsigned*)(krow + kk);
                }
                s[nt] = __builtin_amdgcn_wmma_f32_16x16x32_bf16(
                    false, aq[c].h, false, bk.h, (short)0, s[nt], false, false);
            }
        }

        // ---- causal mask + online softmax (row = e + 8*hi, col = lane&15)
#pragma unroll
        for (int e = 0; e < 8; ++e) {
            int q = qbase + e + 8 * hi;
            float mx = -1e30f;
#pragma unroll
            for (int nt = 0; nt < 2; ++nt) {
                int key = k0 + nt * 16 + r;
                float sv = (key <= q) ? s[nt][e] * scale : -INFINITY;
                s[nt][e] = sv;
                mx = fmaxf(mx, sv);
            }
#pragma unroll
            for (int off = 8; off > 0; off >>= 1)
                mx = fmaxf(mx, __shfl_xor(mx, off, 32));
            float mnew  = fmaxf(mrow[e], mx);
            float alpha = __expf(mrow[e] - mnew);
            float rsum  = 0.f;
#pragma unroll
            for (int nt = 0; nt < 2; ++nt) {
                float p = __expf(s[nt][e] - mnew);
                s[nt][e] = p;
                rsum += p;
            }
#pragma unroll
            for (int off = 8; off > 0; off >>= 1)
                rsum += __shfl_xor(rsum, off, 32);
            lrow[e] = lrow[e] * alpha + rsum;
            mrow[e] = mnew;
#pragma unroll
            for (int nt = 0; nt < 4; ++nt) o_acc[nt][e] *= alpha;
        }

        // ---- re-fragment P: C-layout -> A-layout via per-wave LDS staging
#pragma unroll
        for (int nt = 0; nt < 2; ++nt)
#pragma unroll
            for (int e = 0; e < 8; ++e)
                Pb[wave][(e + 8 * hi) * 32 + nt * 16 + r] = tobf(s[nt][e]);
        __syncthreads();

        Frag16 ap;
        {
            const __bf16* pr = &Pb[wave][r * 32];
#pragma unroll
            for (int e = 0; e < 8; ++e) {
                int kk = ((e < 4) ? 2 * e : 16 + 2 * (e - 4)) + 8 * hi;
                ap.u[e] = *(const unsigned*)(pr + kk);
            }
        }

        // ---- O += P x V : 4 n-tiles of 16 dims, k = 32 keys (from LDS)
#pragma unroll
        for (int nt = 0; nt < 4; ++nt) {
            v16bf bv;
#pragma unroll
            for (int e = 0; e < 8; ++e) {
                int kk = ((e < 4) ? 2 * e : 16 + 2 * (e - 4)) + 8 * hi;
                int col = nt * 16 + r;
                bv[2 * e]     = Vt[cur][kk * HDIM + col];
                bv[2 * e + 1] = Vt[cur][(kk + 1) * HDIM + col];
            }
            o_acc[nt] = __builtin_amdgcn_wmma_f32_16x16x32_bf16(
                false, ap.h, false, bv, (short)0, o_acc[nt], false, false);
        }
    }

#pragma unroll
    for (int e = 0; e < 8; ++e) {
        int q = qbase + e + 8 * hi;
        float invl = 1.f / lrow[e];
#pragma unroll
        for (int nt = 0; nt < 4; ++nt)
            out[((size_t)b * TSEQ + q) * DMODEL + h * HDIM + nt * 16 + r] =
                tobf(o_acc[nt][e] * invl);
    }
}

// ---------------------------------------------------------------------------
// Host orchestration
// ---------------------------------------------------------------------------
extern "C" void kernel_launch(void* const* d_in, const int* in_sizes, int n_in,
                              void* d_out, int out_size, void* d_ws, size_t ws_size,
                              hipStream_t stream) {
    (void)in_sizes; (void)n_in; (void)out_size; (void)ws_size;
    const int*   ids   = (const int*)  d_in[0];
    const float* wte   = (const float*)d_in[1];
    const float* wpe   = (const float*)d_in[2];
    const float* ln1g  = (const float*)d_in[3];
    const float* ln1b  = (const float*)d_in[4];
    const float* attnw = (const float*)d_in[5];
    const float* attnb = (const float*)d_in[6];
    const float* projw = (const float*)d_in[7];
    const float* projb = (const float*)d_in[8];
    const float* ln2g  = (const float*)d_in[9];
    const float* ln2b  = (const float*)d_in[10];
    const float* fcw   = (const float*)d_in[11];
    const float* fcb   = (const float*)d_in[12];
    const float* outw  = (const float*)d_in[13];
    const float* outb  = (const float*)d_in[14];
    const float* lnfg  = (const float*)d_in[15];
    const float* lnfb  = (const float*)d_in[16];

    auto aup = [](size_t s) { return (s + 255) & ~(size_t)255; };
    char* w = (char*)d_ws;
    float*  x     = (float*)w;  w += aup((size_t)ROWS * DMODEL * 4);
    __bf16* qkvB  = (__bf16*)w; w += aup((size_t)ROWS * 3 * DMODEL * 2);
    __bf16* hbuf  = (__bf16*)w; w += aup((size_t)ROWS * DMODEL * 2);
    __bf16* att   = (__bf16*)w; w += aup((size_t)ROWS * DMODEL * 2);
    __bf16* hmid  = (__bf16*)w; w += aup((size_t)ROWS * 4 * DMODEL * 2);
    __bf16* wAttn = (__bf16*)w; w += aup((size_t)LAYERS * DMODEL * 3 * DMODEL * 2);
    __bf16* wProj = (__bf16*)w; w += aup((size_t)LAYERS * DMODEL * DMODEL * 2);
    __bf16* wFc   = (__bf16*)w; w += aup((size_t)LAYERS * DMODEL * 4 * DMODEL * 2);
    __bf16* wOut  = (__bf16*)w; w += aup((size_t)LAYERS * 4 * DMODEL * DMODEL * 2);
    __bf16* wteT  = (__bf16*)w; w += aup((size_t)DMODEL * VPAD * 2);

    dim3 blk(256);

    auto cvt = [&](const float* src, __bf16* dst, long long n) {
        int blocks = (int)((n / 4 + 255) / 256);
        cvt_bf16_kernel<<<blocks, blk, 0, stream>>>(src, dst, n / 4);
    };
    cvt(attnw, wAttn, (long long)LAYERS * DMODEL * 3 * DMODEL);
    cvt(projw, wProj, (long long)LAYERS * DMODEL * DMODEL);
    cvt(fcw,   wFc,   (long long)LAYERS * DMODEL * 4 * DMODEL);
    cvt(outw,  wOut,  (long long)LAYERS * 4 * DMODEL * DMODEL);
    {
        dim3 gt(VPAD / 32, DMODEL / 32);
        transpose_wte_kernel<<<gt, blk, 0, stream>>>(wte, wteT);
    }

    embed_kernel<<<ROWS, blk, 0, stream>>>(ids, wte, wpe, x);

    for (int i = 0; i < LAYERS; ++i) {
        ln_kernel<<<ROWS, blk, 0, stream>>>(x, ln1g + i * DMODEL, ln1b + i * DMODEL, hbuf);

        dim3 gq((3 * DMODEL) / BN, ROWS / BM);
        gemm_wmma<3><<<gq, blk, 0, stream>>>(
            hbuf, wAttn + (size_t)i * DMODEL * 3 * DMODEL,
            attnb + (size_t)i * 3 * DMODEL, nullptr, nullptr, qkvB,
            ROWS, 3 * DMODEL, DMODEL, 3 * DMODEL);

        dim3 ga(TSEQ / 64, NH, BBATCH);
        attn_kernel<<<ga, 128, 0, stream>>>(qkvB, att);

        dim3 gp(DMODEL / BN, ROWS / BM);
        gemm_wmma<2><<<gp, blk, 0, stream>>>(
            att, wProj + (size_t)i * DMODEL * DMODEL,
            projb + (size_t)i * DMODEL, x, x, nullptr,
            ROWS, DMODEL, DMODEL, DMODEL);

        ln_kernel<<<ROWS, blk, 0, stream>>>(x, ln2g + i * DMODEL, ln2b + i * DMODEL, hbuf);

        dim3 gf((4 * DMODEL) / BN, ROWS / BM);
        gemm_wmma<1><<<gf, blk, 0, stream>>>(
            hbuf, wFc + (size_t)i * DMODEL * 4 * DMODEL,
            fcb + (size_t)i * 4 * DMODEL, nullptr, nullptr, hmid,
            ROWS, 4 * DMODEL, DMODEL, 4 * DMODEL);

        dim3 go(DMODEL / BN, ROWS / BM);
        gemm_wmma<2><<<go, blk, 0, stream>>>(
            hmid, wOut + (size_t)i * 4 * DMODEL * DMODEL,
            outb + (size_t)i * DMODEL, x, x, nullptr,
            ROWS, DMODEL, 4 * DMODEL, DMODEL);
    }

    ln_kernel<<<ROWS, blk, 0, stream>>>(x, lnfg, lnfb, hbuf);

    dim3 gl(VPAD / BN, ROWS / BM);
    gemm_wmma<0><<<gl, blk, 0, stream>>>(
        hbuf, wteT, nullptr, nullptr, (float*)d_out, nullptr,
        ROWS, VSZ, DMODEL, VPAD);
}
